// MoELayer_36249523978261
// MI455X (gfx1250) — matrix operand order repaired and checked by reading, not compiled, not run on hardware
//
#include <hip/hip_runtime.h>
#include <cstdint>
#include <cstddef>

// ---------------- problem constants ----------------
constexpr int Tt = 4096;   // tokens (B*S)
constexpr int Dm = 768;    // input dim
constexpr int Ee = 8;      // experts
constexpr int Hh = 3072;   // hidden
constexpr int Oo = 768;    // output dim
constexpr int Pp = 2 * Tt; // (token, k) pairs, K=2

// ---------------- workspace layout (bytes) ----------------
constexpr size_t WS_TOPI   = 0;                            // int[Pp]
constexpr size_t WS_TOPW   = WS_TOPI + (size_t)Pp * 4;     // float[Pp]
constexpr size_t WS_COUNTS = WS_TOPW + (size_t)Pp * 4;     // int[Ee]
constexpr size_t WS_OFFS   = WS_COUNTS + 256;              // int[Ee]
constexpr size_t WS_CURSOR = WS_OFFS + 256;                // int[Ee]
constexpr size_t WS_LIST   = WS_CURSOR + 256;              // int[Pp] pair id per slot
constexpr size_t WS_XB     = WS_LIST + (size_t)Pp * 4;     // bf16[Tt*Dm]
constexpr size_t WS_HB     = WS_XB + (size_t)Tt * Dm * 2;  // bf16[Pp*Hh]
constexpr size_t WS_YB     = WS_HB + (size_t)Pp * Hh * 2;  // float[Pp*Oo]
constexpr size_t WS_W1B    = WS_YB + (size_t)Pp * Oo * 4;  // bf16[Ee*Dm*Hh]
constexpr size_t WS_W2B    = WS_W1B + (size_t)Ee * Dm * Hh * 2; // bf16[Ee*Hh*Oo]

// ---------------- WMMA fragment types ----------------
typedef __attribute__((ext_vector_type(16))) __bf16 v16bf;
typedef __attribute__((ext_vector_type(8)))  float  v8f;
typedef int v4i __attribute__((vector_size(16)));   // matches builtin prototype

union ABfrag { struct { uint4 lo, hi; } q; v16bf v; };
union BPack  { uint4 q; __bf16 h[8]; };

// ---------------- async copy helpers (guarded; fallback = reg+ds path) ----------
#if defined(__has_builtin)
#if __has_builtin(__builtin_amdgcn_global_load_async_to_lds_b128)
#define MOE_HAVE_ASYNC_LDS 1
#endif
#if __has_builtin(__builtin_amdgcn_s_wait_asynccnt)
#define MOE_HAVE_WAIT_ASYNC 1
#endif
#endif

__device__ __forceinline__ void copy16_g2l(const __bf16* g, __bf16* l) {
#if defined(MOE_HAVE_ASYNC_LDS)
    __builtin_amdgcn_global_load_async_to_lds_b128(
        (__attribute__((address_space(1))) v4i*)(uintptr_t)g,
        (__attribute__((address_space(3))) v4i*)(uint32_t)(uintptr_t)l,
        0, 0);
#else
    *(uint4*)l = *(const uint4*)g;
#endif
}

__device__ __forceinline__ void wait_async_copies() {
#if defined(MOE_HAVE_ASYNC_LDS) && defined(MOE_HAVE_WAIT_ASYNC)
    __builtin_amdgcn_s_wait_asynccnt(0);
#endif
}

// ---------------- small kernels ----------------
__global__ void zero_counts_kernel(int* counts) {
    if (threadIdx.x < Ee) counts[threadIdx.x] = 0;
}

// fp32 -> bf16 streaming convert, 8 elements/thread (exact multiple grids)
__global__ void conv_kernel(const float* __restrict__ src, __bf16* __restrict__ dst) {
    int gid = blockIdx.x * blockDim.x + threadIdx.x;
    size_t idx = (size_t)gid * 8;
    float4 a = *(const float4*)(src + idx);
    float4 b = *(const float4*)(src + idx + 4);
    alignas(16) __bf16 t[8];
    t[0] = (__bf16)a.x; t[1] = (__bf16)a.y; t[2] = (__bf16)a.z; t[3] = (__bf16)a.w;
    t[4] = (__bf16)b.x; t[5] = (__bf16)b.y; t[6] = (__bf16)b.z; t[7] = (__bf16)b.w;
    *(uint4*)(dst + idx) = *(const uint4*)t;
}

__global__ __launch_bounds__(256)
void gate_kernel(const float* __restrict__ x, const float* __restrict__ Wg,
                 int* __restrict__ topi, float* __restrict__ topw,
                 int* __restrict__ counts, float* __restrict__ gate_out) {
    const int lane = threadIdx.x & 31;
    const int wave = threadIdx.x >> 5;
    const int t = blockIdx.x * 8 + wave;              // one wave32 per token

    float xr[24];
#pragma unroll
    for (int i = 0; i < 24; ++i) xr[i] = x[(size_t)t * Dm + lane + i * 32];

    float lg[8];
#pragma unroll
    for (int e = 0; e < 8; ++e) {
        float a = 0.f;
#pragma unroll
        for (int i = 0; i < 24; ++i) a += xr[i] * Wg[(lane + i * 32) * Ee + e];
        lg[e] = a;
    }
#pragma unroll
    for (int off = 16; off > 0; off >>= 1) {
#pragma unroll
        for (int e = 0; e < 8; ++e) lg[e] += __shfl_xor(lg[e], off, 32);
    }
    if (lane == 0) {
        int e0 = 0; float v0 = lg[0];
#pragma unroll
        for (int e = 1; e < 8; ++e) if (lg[e] > v0) { v0 = lg[e]; e0 = e; }
        int e1 = (e0 == 0) ? 1 : 0; float v1 = lg[e1];
#pragma unroll
        for (int e = 0; e < 8; ++e) if (e != e0 && lg[e] > v1) { v1 = lg[e]; e1 = e; }
        float b  = __expf(v1 - v0);                   // softmax over {v0, v1}
        float s  = 1.f + b;
        float w0 = 1.f / s, w1 = b / s;
#pragma unroll
        for (int e = 0; e < 8; ++e)
            gate_out[(size_t)t * Ee + e] = (e == e0) ? w0 : ((e == e1) ? w1 : 0.f);
        topi[2 * t] = e0; topi[2 * t + 1] = e1;
        topw[2 * t] = w0; topw[2 * t + 1] = w1;
        atomicAdd(&counts[e0], 1);
        atomicAdd(&counts[e1], 1);
    }
}

__global__ void scan_kernel(const int* __restrict__ counts,
                            int* __restrict__ offs, int* __restrict__ cursor) {
    if (threadIdx.x == 0 && blockIdx.x == 0) {
        int run = 0;
        for (int e = 0; e < Ee; ++e) { offs[e] = run; cursor[e] = run; run += counts[e]; }
    }
}

__global__ void place_kernel(const int* __restrict__ topi,
                             int* __restrict__ cursor, int* __restrict__ listpair) {
    int p = blockIdx.x * blockDim.x + threadIdx.x;
    if (p >= Pp) return;
    int e = topi[p];
    int slot = atomicAdd(&cursor[e], 1);
    listpair[slot] = p;
}

// ---------------- GEMM1: h[slot] = gelu(x[tok] @ W1[e] + b1[e]) ----------------
// block = 256 threads (8 waves). tile: 128 rows x 64 cols x 32 k (bf16 WMMA 16x16x32)
__global__ __launch_bounds__(256)
void gemm1_kernel(const __bf16* __restrict__ xb, const __bf16* __restrict__ W1b,
                  const float* __restrict__ b1, const int* __restrict__ listpair,
                  const int* __restrict__ counts, const int* __restrict__ offs,
                  __bf16* __restrict__ hb) {
    const int e = blockIdx.z;
    const int cnt = counts[e];
    const int row0 = blockIdx.x * 128;
    if (row0 >= cnt) return;
    const int base = offs[e];
    const int n0 = blockIdx.y * 64;
    const int tid = threadIdx.x;
    const int lane = tid & 31, wave = tid >> 5;

    __shared__ alignas(16) __bf16 As[128][32];   // [row][k]      8 KB
    __shared__ alignas(16) __bf16 Bs[64][40];    // [n][k] padded 5 KB
    __shared__ int tokS[128];

    if (tid < 128) {
        int r = (row0 + tid < cnt) ? (row0 + tid) : (cnt - 1); // clamp: always valid
        tokS[tid] = listpair[base + r] >> 1;
    }
    __syncthreads();

    v8f acc[4] = {};

    // A staging: 2 threads per row, 16 bf16 (2 x b128) each -- loop-invariant base
    const int ar  = tid >> 1;
    const int ac0 = (tid & 1) * 16;
    const __bf16* a0 = xb + (size_t)tokS[ar] * Dm + ac0;
    // B staging: thread -> one k-row (32 rows x 8 threads), 8 bf16 along n
    const int bk  = tid >> 3;
    const int bn8 = (tid & 7) * 8;
    const __bf16* bsrc0 = W1b + (size_t)e * Dm * Hh + (size_t)bk * Hh + n0 + bn8;

    for (int kk = 0; kk < Dm; kk += 32) {
        copy16_g2l(a0 + kk,     &As[ar][ac0]);
        copy16_g2l(a0 + kk + 8, &As[ar][ac0 + 8]);
        BPack w; w.q = *(const uint4*)(bsrc0 + (size_t)kk * Hh);
#pragma unroll
        for (int i = 0; i < 8; ++i) Bs[bn8 + i][bk] = w.h[i];   // transpose on store
        if (kk + 32 < Dm)
            __builtin_prefetch(bsrc0 + (size_t)(kk + 32) * Hh, 0, 0);
        wait_async_copies();
        __syncthreads();

        // A frag (16x32): lanes 0-15 K{0..7,16..23}, lanes 16-31 K{8..15,24..31}
        const int m  = wave * 16 + (lane & 15);
        const int kb = (lane >> 4) * 8;
        ABfrag af;
        af.q.lo = *(const uint4*)&As[m][kb];
        af.q.hi = *(const uint4*)&As[m][kb + 16];
        const int kb2 = (lane >> 4) * 8;          // B: lanes 0-15 K{0..7,16..23}, 16-31 K{8..15,24..31}
#pragma unroll
        for (int tn = 0; tn < 4; ++tn) {
            ABfrag bf;
            bf.q.lo = *(const uint4*)&Bs[tn * 16 + (lane & 15)][kb2];
            bf.q.hi = *(const uint4*)&Bs[tn * 16 + (lane & 15)][kb2 + 16];
            acc[tn] = __builtin_amdgcn_wmma_f32_16x16x32_bf16(
                false, af.v, false, bf.v, (short)0, acc[tn], false, false);
        }
        __syncthreads();
    }

    // ---- epilogue: bias + exact GELU -> bf16 h ----
    const int col_l = lane & 15;
    const int rsel  = (lane >> 4) * 8;
#pragma unroll
    for (int tn = 0; tn < 4; ++tn) {
        const int col = n0 + tn * 16 + col_l;
        const float bv = b1[(size_t)e * Hh + col];
#pragma unroll
        for (int r = 0; r < 8; ++r) {
            const int lr = wave * 16 + r + rsel;
            if (row0 + lr < cnt) {
                float v = acc[tn][r] + bv;
                v = 0.5f * v * (1.f + erff(v * 0.70710678118654752f));
                hb[(size_t)(base + row0 + lr) * Hh + col] = (__bf16)v;
            }
        }
    }
}

// ---------------- GEMM2: y[pair] = h[slot] @ W2[e] + b2[e] ----------------
__global__ __launch_bounds__(256)
void gemm2_kernel(const __bf16* __restrict__ hb, const __bf16* __restrict__ W2b,
                  const float* __restrict__ b2, const int* __restrict__ listpair,
                  const int* __restrict__ counts, const int* __restrict__ offs,
                  float* __restrict__ yb) {
    const int e = blockIdx.z;
    const int cnt = counts[e];
    const int row0 = blockIdx.x * 128;
    if (row0 >= cnt) return;
    const int base = offs[e];
    const int n0 = blockIdx.y * 64;
    const int tid = threadIdx.x;
    const int lane = tid & 31, wave = tid >> 5;

    __shared__ alignas(16) __bf16 As[128][32];
    __shared__ alignas(16) __bf16 Bs[64][40];
    __shared__ int pairS[128];

    if (tid < 128) {
        int p = 0;
        if (row0 + tid < cnt) p = listpair[base + row0 + tid];
        pairS[tid] = p;
    }
    __syncthreads();

    v8f acc[4] = {};

    const int ar  = tid >> 1;
    const int ac0 = (tid & 1) * 16;
    const int arow = (row0 + ar < cnt) ? (row0 + ar) : (cnt - 1);   // clamp
    const __bf16* a0 = hb + (size_t)(base + arow) * Hh + ac0;
    const int bk  = tid >> 3;
    const int bn8 = (tid & 7) * 8;
    const __bf16* bsrc0 = W2b + (size_t)e * Hh * Oo + (size_t)bk * Oo + n0 + bn8;

    for (int kk = 0; kk < Hh; kk += 32) {
        copy16_g2l(a0 + kk,     &As[ar][ac0]);
        copy16_g2l(a0 + kk + 8, &As[ar][ac0 + 8]);
        BPack w; w.q = *(const uint4*)(bsrc0 + (size_t)kk * Oo);
#pragma unroll
        for (int i = 0; i < 8; ++i) Bs[bn8 + i][bk] = w.h[i];
        if (kk + 32 < Hh)
            __builtin_prefetch(bsrc0 + (size_t)(kk + 32) * Oo, 0, 0);
        wait_async_copies();
        __syncthreads();

        const int m  = wave * 16 + (lane & 15);
        const int kb = (lane >> 4) * 8;
        ABfrag af;
        af.q.lo = *(const uint4*)&As[m][kb];
        af.q.hi = *(const uint4*)&As[m][kb + 16];
        const int kb2 = (lane >> 4) * 8;
#pragma unroll
        for (int tn = 0; tn < 4; ++tn) {
            ABfrag bf;
            bf.q.lo = *(const uint4*)&Bs[tn * 16 + (lane & 15)][kb2];
            bf.q.hi = *(const uint4*)&Bs[tn * 16 + (lane & 15)][kb2 + 16];
            acc[tn] = __builtin_amdgcn_wmma_f32_16x16x32_bf16(
                false, af.v, false, bf.v, (short)0, acc[tn], false, false);
        }
        __syncthreads();
    }

    const int col_l = lane & 15;
    const int rsel  = (lane >> 4) * 8;
#pragma unroll
    for (int tn = 0; tn < 4; ++tn) {
        const int col = n0 + tn * 16 + col_l;
        const float bv = b2[(size_t)e * Oo + col];
#pragma unroll
        for (int r = 0; r < 8; ++r) {
            const int lr = wave * 16 + r + rsel;
            if (row0 + lr < cnt)
                yb[(size_t)pairS[lr] * Oo + col] = acc[tn][r] + bv;
        }
    }
}

// ---------------- combine: out[t] = w0*y[2t] + w1*y[2t+1] ----------------
__global__ __launch_bounds__(256)
void combine_kernel(const float* __restrict__ yb, const float* __restrict__ topw,
                    float* __restrict__ out) {
    constexpr int perRow = Oo / 4;   // 192 float4 per row
    int gid = blockIdx.x * blockDim.x + threadIdx.x;
    if (gid >= Tt * perRow) return;
    int t = gid / perRow;
    int c = (gid % perRow) * 4;
    float w0 = topw[2 * t], w1 = topw[2 * t + 1];
    float4 y0 = *(const float4*)(yb + (size_t)(2 * t) * Oo + c);
    float4 y1 = *(const float4*)(yb + (size_t)(2 * t + 1) * Oo + c);
    float4 o;
    o.x = w0 * y0.x + w1 * y1.x;
    o.y = w0 * y0.y + w1 * y1.y;
    o.z = w0 * y0.z + w1 * y1.z;
    o.w = w0 * y0.w + w1 * y1.w;
    *(float4*)(out + (size_t)t * Oo + c) = o;
}

// ---------------- host launcher ----------------
extern "C" void kernel_launch(void* const* d_in, const int* in_sizes, int n_in,
                              void* d_out, int out_size, void* d_ws, size_t ws_size,
                              hipStream_t stream) {
    (void)in_sizes; (void)n_in; (void)out_size; (void)ws_size;
    const float* x  = (const float*)d_in[0];
    const float* Wg = (const float*)d_in[1];
    const float* W1 = (const float*)d_in[2];
    const float* b1 = (const float*)d_in[3];
    const float* W2 = (const float*)d_in[4];
    const float* b2 = (const float*)d_in[5];

    float* out      = (float*)d_out;
    float* gate_out = out + (size_t)Tt * Oo;   // second output section

    char* ws = (char*)d_ws;
    int*    topi     = (int*)   (ws + WS_TOPI);
    float*  topw     = (float*) (ws + WS_TOPW);
    int*    counts   = (int*)   (ws + WS_COUNTS);
    int*    offs     = (int*)   (ws + WS_OFFS);
    int*    cursor   = (int*)   (ws + WS_CURSOR);
    int*    listpair = (int*)   (ws + WS_LIST);
    __bf16* xb       = (__bf16*)(ws + WS_XB);
    __bf16* hb       = (__bf16*)(ws + WS_HB);
    float*  yb       = (float*) (ws + WS_YB);
    __bf16* W1b      = (__bf16*)(ws + WS_W1B);
    __bf16* W2b      = (__bf16*)(ws + WS_W2B);

    zero_counts_kernel<<<1, 32, 0, stream>>>(counts);
    conv_kernel<<<Tt * Dm / 8 / 256, 256, 0, stream>>>(x, xb);
    conv_kernel<<<(int)((size_t)Ee * Dm * Hh / 8 / 256), 256, 0, stream>>>(W1, W1b);
    conv_kernel<<<(int)((size_t)Ee * Hh * Oo / 8 / 256), 256, 0, stream>>>(W2, W2b);
    gate_kernel<<<Tt / 8, 256, 0, stream>>>(x, Wg, topi, topw, counts, gate_out);
    scan_kernel<<<1, 32, 0, stream>>>(counts, offs, cursor);
    place_kernel<<<(Pp + 255) / 256, 256, 0, stream>>>(topi, cursor, listpair);
    gemm1_kernel<<<dim3(Tt / 128, Hh / 64, Ee), 256, 0, stream>>>(
        xb, W1b, b1, listpair, counts, offs, hb);
    gemm2_kernel<<<dim3(Tt / 128, Oo / 64, Ee), 256, 0, stream>>>(
        hb, W2b, b2, listpair, counts, offs, yb);
    combine_kernel<<<(Tt * (Oo / 4) + 255) / 256, 256, 0, stream>>>(yb, topw, out);
}
